// DeepSetsEnsemble2_58145267253429
// MI455X (gfx1250) — matrix operand order repaired and checked by reading, not compiled, not run on hardware
//
#include <hip/hip_runtime.h>
#include <hip/hip_bf16.h>
#include <math.h>

// ---------------------------------------------------------------------------
// CDNA5 (gfx1250) fused DeepSets implementation.
// All GEMMs go through v_wmma_f32_16x16x32_f16 (wave32 WMMA).
// ---------------------------------------------------------------------------

typedef __attribute__((ext_vector_type(16))) _Float16 v16h;
typedef __attribute__((ext_vector_type(8)))  float    v8f;
typedef __attribute__((ext_vector_type(2)))  _Float16 h2v;

#define NEV   4096
#define PTS   64
#define NTOT  (NEV * PTS)
#define ZLD   288          // 260 padded to 9*32
#define BNEPS 1e-5f

__device__ __forceinline__ v8f wmma_f16(v16h a, v16h b, v8f c) {
  return __builtin_amdgcn_wmma_f32_16x16x32_f16(false, a, false, b,
                                                (short)0, c, false, false);
}

// ---- A fragment (16x32 f16), source row-major [M][ld], pairs contiguous ----
// Contiguous-k order used here (matches B): lane m = lane&15 ; VGPR v holds K = (lane>=16?16:0) + 2v, 2v+1
__device__ __forceinline__ v16h frag_a_rm(const _Float16* __restrict__ p,
                                          int row0, int k0, int ld) {
  const int lane = threadIdx.x & 31;
  const int m    = lane & 15;
  const int kh   = (lane & 16) ? 16 : 0;
  const _Float16* r = p + (long)(row0 + m) * ld + k0;
  v16h a;
#pragma unroll
  for (int v = 0; v < 8; ++v) {
    const int k = kh + (v << 1);
    h2v pr = *(const h2v*)(r + k);
    a[2 * v]     = pr[0];
    a[2 * v + 1] = pr[1];
  }
  return a;
}

// ---- B fragment (32x16 f16), source "event-major" [n][ld] so K contiguous ----
// Contiguous-k order used here (matches A): lane n = lane&15 ; VGPR v holds K = (lane>=16?16:0) + 2v, 2v+1
__device__ __forceinline__ v16h frag_b_cm(const _Float16* __restrict__ p,
                                          int k0, int n0, int ld) {
  const int lane = threadIdx.x & 31;
  const int n    = lane & 15;
  const int kh   = (lane & 16) ? 16 : 0;
  const _Float16* c = p + (long)(n0 + n) * ld + k0 + kh;
  v16h b;
#pragma unroll
  for (int v = 0; v < 8; ++v) {
    h2v pb = *(const h2v*)(c + 2 * v);
    b[2 * v]     = pb[0];
    b[2 * v + 1] = pb[1];
  }
  return b;
}

// ---- D stores.  D layout: lane n = lane&15 ; VGPR r -> m = r + (lane>=16?8:0)
__device__ __forceinline__ void store_d_act(_Float16* __restrict__ p, v8f d,
                                            const float* __restrict__ bias,
                                            int m0, int n0, int ld) {
  const int lane = threadIdx.x & 31;
  const int n    = lane & 15;
  const int mo   = (lane & 16) ? 8 : 0;
#pragma unroll
  for (int r = 0; r < 8; ++r) {
    const int m = m0 + mo + r;
    float v = d[r] + bias[m];
    v = v > 0.f ? v : 0.f;
    p[(long)(n0 + n) * ld + m] = (_Float16)v;
  }
}

__device__ __forceinline__ void store_d_pool(float* __restrict__ sp, v8f d,
                                             const float* __restrict__ bias,
                                             int m0, int n0) {
  const int lane = threadIdx.x & 31;
  const int n    = lane & 15;
  const int mo   = (lane & 16) ? 8 : 0;
#pragma unroll
  for (int r = 0; r < 8; ++r) {
    const int m = m0 + mo + r;
    float v = d[r] + bias[m];
    v = v > 0.f ? v : 0.f;
    sp[m * PTS + n0 + n] = v;
  }
}

__device__ __forceinline__ void store_d_f32(float* __restrict__ p, v8f d,
                                            int m0, int n0, int stride) {
  const int lane = threadIdx.x & 31;
  const int n    = lane & 15;
  const int mo   = (lane & 16) ? 8 : 0;
#pragma unroll
  for (int r = 0; r < 8; ++r)
    p[(long)(m0 + mo + r) * stride + (n0 + n)] = d[r];
}

// ---------------------------------------------------------------------------
// Weight convert + pad (f32 [M][Kin] -> f16 [M][Kp], zero-padded K)
// ---------------------------------------------------------------------------
__global__ void convert_pad(const float* __restrict__ src,
                            _Float16* __restrict__ dst,
                            int M, int Kin, int Kp) {
  const int i = blockIdx.x * 256 + threadIdx.x;
  if (i >= M * Kp) return;
  const int m = i / Kp, k = i % Kp;
  dst[i] = (k < Kin) ? (_Float16)src[m * Kin + k] : (_Float16)0.f;
}

// z columns 256..287: met passthrough + zero padding
__global__ void assemble_z(const float* __restrict__ x2, _Float16* __restrict__ z) {
  const int i = blockIdx.x * 256 + threadIdx.x;
  if (i >= NEV * 32) return;
  const int e = i >> 5, c = 256 + (i & 31);
  z[(long)e * ZLD + c] = (c < 260) ? (_Float16)x2[e * 4 + (c - 256)] : (_Float16)0.f;
}

// ---------------------------------------------------------------------------
// Fused phi stack + segment-mean pool. 1 block = 1 event (64 points), 8 waves.
// LDS (64 KB total, regions aliased across phases):
//   [0,32K)   sH1 (layer1 out, point-major [64][256] f16) -> later sPool f32 [128][64]
//   [32K,64K) sH2 (layer2 out) ; its first 4 KB doubles as sX [64][32] in phase 1
// ---------------------------------------------------------------------------
__global__ __launch_bounds__(256) void phi_kernel(
    const float* __restrict__ x, int Cin,
    const _Float16* __restrict__ w1f, const float* __restrict__ b1,
    const _Float16* __restrict__ w2f, const float* __restrict__ b2,
    const _Float16* __restrict__ w3f, const float* __restrict__ b3,
    _Float16* __restrict__ zbuf, int zcol0) {
  __shared__ __align__(16) char smem[65536];
  _Float16* sH1  = (_Float16*)smem;
  _Float16* sH2  = (_Float16*)(smem + 32768);
  _Float16* sX   = sH2;                 // alias: dead before phase 2 writes
  float*    sPool = (float*)smem;       // alias: sH1 dead in phase 3

  const int e    = blockIdx.x;
  const int tid  = threadIdx.x;
  const int wave = tid >> 5;
  const long col0 = (long)e * PTS;

  // ---- stage input tile, zero-pad K to 32 ----
  for (int i = tid; i < PTS * 32; i += 256) sX[i] = (_Float16)0.f;
  __syncthreads();
  for (int i = tid; i < PTS * Cin; i += 256) {
    const int k = i / PTS, n = i % PTS;     // consecutive tids -> coalesced n
    sX[n * 32 + k] = (_Float16)x[(long)k * NTOT + col0 + n];
  }
  __syncthreads();

  // ---- layer 1: [256 x 64] = W1[256x32] @ X[32x64] ----
  {
    const int m0 = wave * 32;
#pragma unroll
    for (int nt = 0; nt < 4; ++nt) {
      v16h b = frag_b_cm(sX, 0, nt * 16, 32);
#pragma unroll
      for (int mi = 0; mi < 2; ++mi) {
        v16h a = frag_a_rm(w1f, m0 + mi * 16, 0, 32);
        v8f acc = {};
        acc = wmma_f16(a, b, acc);
        store_d_act(sH1, acc, b1, m0 + mi * 16, nt * 16, 256);
      }
    }
  }
  __syncthreads();

  // ---- layer 2: [256 x 64] = W2[256x256] @ H1 ----
  {
    const int m0 = wave * 32;
#pragma unroll
    for (int nt = 0; nt < 4; ++nt) {
      v8f acc0 = {}, acc1 = {};
      for (int k0 = 0; k0 < 256; k0 += 32) {
        v16h b = frag_b_cm(sH1, k0, nt * 16, 256);
        acc0 = wmma_f16(frag_a_rm(w2f, m0, k0, 256), b, acc0);
        acc1 = wmma_f16(frag_a_rm(w2f, m0 + 16, k0, 256), b, acc1);
      }
      store_d_act(sH2, acc0, b2, m0, nt * 16, 256);
      store_d_act(sH2, acc1, b2, m0 + 16, nt * 16, 256);
    }
  }
  __syncthreads();

  // ---- layer 3: [128 x 64] = W3[128x256] @ H2, result into f32 pool buffer ----
  {
    const int m0 = (wave & 3) * 32;
    const int ntb = (wave >> 2) * 2;
#pragma unroll
    for (int t = 0; t < 2; ++t) {
      const int n0 = (ntb + t) * 16;
      v8f acc0 = {}, acc1 = {};
      for (int k0 = 0; k0 < 256; k0 += 32) {
        v16h b = frag_b_cm(sH2, k0, n0, 256);
        acc0 = wmma_f16(frag_a_rm(w3f, m0, k0, 256), b, acc0);
        acc1 = wmma_f16(frag_a_rm(w3f, m0 + 16, k0, 256), b, acc1);
      }
      store_d_pool(sPool, acc0, b3, m0, n0);
      store_d_pool(sPool, acc1, b3, m0 + 16, n0);
    }
  }
  __syncthreads();

  // ---- segment mean over the 64 points of this event ----
  if (tid < 128) {
    float s = 0.f;
#pragma unroll 8
    for (int j = 0; j < PTS; ++j) s += sPool[tid * PTS + j];
    zbuf[(long)e * ZLD + zcol0 + tid] = (_Float16)(s * (1.f / PTS));
  }
}

// ---------------------------------------------------------------------------
// MLP GEMM: Hout[M][4096] (f32, channel-major) = Wf[M][Kp] @ Ain^T
// Ain is event-major f16 [4096][Kp].  Block = 64 events, wave owns M/8 rows.
// Linear bias is skipped: BatchNorm subtracts the batch mean, so it cancels.
// ---------------------------------------------------------------------------
template <int NMT>
__global__ __launch_bounds__(256) void mlp_gemm(
    const _Float16* __restrict__ Wf, int Kp,
    const _Float16* __restrict__ Ain,
    float* __restrict__ Hout) {
  const int wave = threadIdx.x >> 5;
  const int m0w  = wave * 16 * NMT;
  const int e0   = blockIdx.x * 64;
#pragma unroll
  for (int nt = 0; nt < 4; ++nt) {
    const int n0 = e0 + nt * 16;
    v8f acc[NMT] = {};
    for (int k0 = 0; k0 < Kp; k0 += 32) {
      v16h b = frag_b_cm(Ain, k0, n0, Kp);
#pragma unroll
      for (int mi = 0; mi < NMT; ++mi)
        acc[mi] = wmma_f16(frag_a_rm(Wf, m0w + mi * 16, k0, Kp), b, acc[mi]);
    }
#pragma unroll
    for (int mi = 0; mi < NMT; ++mi)
      store_d_f32(Hout, acc[mi], m0w + mi * 16, n0, NEV);
  }
}

// ---- BatchNorm batch statistics: one block per channel row of H[M][4096] ----
__global__ __launch_bounds__(256) void bn_stats(const float* __restrict__ H,
                                                float* __restrict__ mu,
                                                float* __restrict__ rs) {
  __shared__ float ss[256], sq[256];
  const int r = blockIdx.x;
  const float* row = H + (long)r * NEV;
  float s = 0.f, q = 0.f;
  for (int i = threadIdx.x; i < NEV; i += 256) {
    const float v = row[i];
    s += v; q += v * v;
  }
  ss[threadIdx.x] = s; sq[threadIdx.x] = q;
  __syncthreads();
  for (int st = 128; st > 0; st >>= 1) {
    if (threadIdx.x < st) {
      ss[threadIdx.x] += ss[threadIdx.x + st];
      sq[threadIdx.x] += sq[threadIdx.x + st];
    }
    __syncthreads();
  }
  if (threadIdx.x == 0) {
    const float m = ss[0] * (1.f / NEV);
    const float var = sq[0] * (1.f / NEV) - m * m;   // biased, as torch BN fwd
    mu[r] = m;
    rs[r] = rsqrtf(var + BNEPS);
  }
}

// ---- Apply BN + ReLU, write f16 event-major [4096][M] for next GEMM's B ----
__global__ void bn_apply(const float* __restrict__ H, int M,
                         const float* __restrict__ g, const float* __restrict__ be,
                         const float* __restrict__ mu, const float* __restrict__ rs,
                         _Float16* __restrict__ A) {
  const int i = blockIdx.x * 256 + threadIdx.x;
  if (i >= M * NEV) return;
  const int m = i / NEV, n = i % NEV;
  float v = (H[i] - mu[m]) * rs[m] * g[m] + be[m];
  v = v > 0.f ? v : 0.f;
  A[(long)n * M + m] = (_Float16)v;
}

// ---- Final 256 -> 1 linear + sigmoid ----
__global__ void final_kernel(const _Float16* __restrict__ a2,
                             const float* __restrict__ w,
                             const float* __restrict__ b,
                             float* __restrict__ out) {
  const int e = blockIdx.x * 256 + threadIdx.x;
  if (e >= NEV) return;
  float acc = b[0];
  const _Float16* r = a2 + (long)e * 256;
#pragma unroll 8
  for (int k = 0; k < 256; ++k) acc += (float)r[k] * w[k];
  out[e] = 1.f / (1.f + expf(-acc));
}

// ---------------------------------------------------------------------------
extern "C" void kernel_launch(void* const* d_in, const int* in_sizes, int n_in,
                              void* d_out, int out_size, void* d_ws, size_t ws_size,
                              hipStream_t stream) {
  (void)in_sizes; (void)n_in; (void)out_size; (void)ws_size;

  const float* x0 = (const float*)d_in[0];
  const float* x1 = (const float*)d_in[1];
  const float* x2 = (const float*)d_in[2];
  const float* p0W1 = (const float*)d_in[3];  const float* p0b1 = (const float*)d_in[4];
  const float* p0W2 = (const float*)d_in[5];  const float* p0b2 = (const float*)d_in[6];
  const float* p0W3 = (const float*)d_in[7];  const float* p0b3 = (const float*)d_in[8];
  const float* p1W1 = (const float*)d_in[9];  const float* p1b1 = (const float*)d_in[10];
  const float* p1W2 = (const float*)d_in[11]; const float* p1b2 = (const float*)d_in[12];
  const float* p1W3 = (const float*)d_in[13]; const float* p1b3 = (const float*)d_in[14];
  const float* mW0 = (const float*)d_in[15];  // b0 (d_in[16]) cancels under BN
  const float* mg0 = (const float*)d_in[17];  const float* mbe0 = (const float*)d_in[18];
  const float* mW1 = (const float*)d_in[19];  // b1 (d_in[20]) cancels under BN
  const float* mg1 = (const float*)d_in[21];  const float* mbe1 = (const float*)d_in[22];
  const float* mW2 = (const float*)d_in[23];  const float* mb2 = (const float*)d_in[24];
  float* out = (float*)d_out;

  // ---- workspace carve-up (256 B aligned) ----
  char* wp = (char*)d_ws;
  size_t off = 0;
  auto carve = [&](size_t bytes) {
    void* p = wp + off;
    off += (bytes + 255) & ~(size_t)255;
    return p;
  };
  _Float16* wphi0_1 = (_Float16*)carve(256 * 32 * 2);
  _Float16* wphi0_2 = (_Float16*)carve(256 * 256 * 2);
  _Float16* wphi0_3 = (_Float16*)carve(128 * 256 * 2);
  _Float16* wphi1_1 = (_Float16*)carve(256 * 32 * 2);
  _Float16* wphi1_2 = (_Float16*)carve(256 * 256 * 2);
  _Float16* wphi1_3 = (_Float16*)carve(128 * 256 * 2);
  _Float16* wmlp0   = (_Float16*)carve((size_t)512 * ZLD * 2);
  _Float16* wmlp1   = (_Float16*)carve((size_t)256 * 512 * 2);
  _Float16* zbuf    = (_Float16*)carve((size_t)NEV * ZLD * 2);
  float*    h1      = (float*)carve((size_t)512 * NEV * 4);
  _Float16* a1      = (_Float16*)carve((size_t)NEV * 512 * 2);
  float*    h2      = (float*)carve((size_t)256 * NEV * 4);
  _Float16* a2      = (_Float16*)carve((size_t)NEV * 256 * 2);
  float*    mu1     = (float*)carve(512 * 4);
  float*    rs1     = (float*)carve(512 * 4);
  float*    mu2     = (float*)carve(256 * 4);
  float*    rs2     = (float*)carve(256 * 4);

  auto cpgrid = [](int n) { return dim3((n + 255) / 256); };

  // ---- weight conversion / padding ----
  convert_pad<<<cpgrid(256 * 32),  256, 0, stream>>>(p0W1, wphi0_1, 256, 16, 32);
  convert_pad<<<cpgrid(256 * 256), 256, 0, stream>>>(p0W2, wphi0_2, 256, 256, 256);
  convert_pad<<<cpgrid(128 * 256), 256, 0, stream>>>(p0W3, wphi0_3, 128, 256, 256);
  convert_pad<<<cpgrid(256 * 32),  256, 0, stream>>>(p1W1, wphi1_1, 256, 8, 32);
  convert_pad<<<cpgrid(256 * 256), 256, 0, stream>>>(p1W2, wphi1_2, 256, 256, 256);
  convert_pad<<<cpgrid(128 * 256), 256, 0, stream>>>(p1W3, wphi1_3, 128, 256, 256);
  convert_pad<<<cpgrid(512 * ZLD), 256, 0, stream>>>(mW0, wmlp0, 512, 260, ZLD);
  convert_pad<<<cpgrid(256 * 512), 256, 0, stream>>>(mW1, wmlp1, 256, 512, 512);

  // ---- z columns 256..287 (met + zero pad) ----
  assemble_z<<<cpgrid(NEV * 32), 256, 0, stream>>>(x2, zbuf);

  // ---- phi stacks + pooling -> z[:,0:128] and z[:,128:256] ----
  phi_kernel<<<NEV, 256, 0, stream>>>(x0, 16, wphi0_1, p0b1, wphi0_2, p0b2,
                                      wphi0_3, p0b3, zbuf, 0);
  phi_kernel<<<NEV, 256, 0, stream>>>(x1, 8, wphi1_1, p1b1, wphi1_2, p1b2,
                                      wphi1_3, p1b3, zbuf, 128);

  // ---- MLP block 1: 260(288) -> 512, BN, ReLU ----
  mlp_gemm<4><<<NEV / 64, 256, 0, stream>>>(wmlp0, ZLD, zbuf, h1);
  bn_stats<<<512, 256, 0, stream>>>(h1, mu1, rs1);
  bn_apply<<<cpgrid(512 * NEV), 256, 0, stream>>>(h1, 512, mg0, mbe0, mu1, rs1, a1);

  // ---- MLP block 2: 512 -> 256, BN, ReLU ----
  mlp_gemm<2><<<NEV / 64, 256, 0, stream>>>(wmlp1, 512, a1, h2);
  bn_stats<<<256, 256, 0, stream>>>(h2, mu2, rs2);
  bn_apply<<<cpgrid(256 * NEV), 256, 0, stream>>>(h2, 256, mg1, mbe1, mu2, rs2, a2);

  // ---- 256 -> 1 + sigmoid ----
  final_kernel<<<cpgrid(NEV), 256, 0, stream>>>(a2, mW2, mb2, out);
}
